// FastWeightTTT_66425964200054
// MI455X (gfx1250) — matrix operand order repaired and checked by reading, not compiled, not run on hardware
//
#include <hip/hip_runtime.h>

// ---------------------------------------------------------------------------
// FastWeightTTT on gfx1250 (MI455X): bf16 WMMA GEMMs + fp32 master state.
// D=1024, H=2048, C=512, B=4, T=4096, 8 chunk steps, momentum=0.9, lr=1.0.
// GEMM core: 256-thread block (8 waves), 128x128 block tile, K-step 32,
// double-buffered LDS staging (padded stride to kill bank conflicts).
// ---------------------------------------------------------------------------

#define DM 1024
#define HH 2048
#define CC 512
#define BB 4
#define TT 4096
#define NCHUNK (TT / CC)
#define MU 0.9f
#define LDSTR 40  // LDS row stride in bf16 (32 data + 8 pad = 80B = 20 banks)

typedef __attribute__((ext_vector_type(16))) __bf16 v16bf;
typedef __attribute__((ext_vector_type(8)))  float  v8f;

// ---------------- WMMA fragment loaders from LDS (wave32) ------------------
// A fragment (16x32, M x K): lane l: m=l&15, g=l>>4;
//   elem i: k = (i&7) + 8*g + 16*(i>>3)  -> two 16B reads at +8g and +16+8g
__device__ __forceinline__ v16bf frag_a_lds(const __bf16* As, int moff) {
  int lane = threadIdx.x & 31;
  int m = lane & 15, g = lane >> 4;
  const __bf16* p = As + (moff + m) * LDSTR + 8 * g;
  uint4 lo = *(const uint4*)(p);
  uint4 hi = *(const uint4*)(p + 16);
  const __bf16* lp = (const __bf16*)&lo;
  const __bf16* hp = (const __bf16*)&hi;
  v16bf r;
#pragma unroll
  for (int i = 0; i < 8; ++i) { r[i] = lp[i]; r[8 + i] = hp[i]; }
  return r;
}

// B fragment (32x16, K x N) staged as [N][K]: lane l: n=l&15, g=l>>4;
//   elem i: k = (i&7) + 8*g + 16*(i>>3)  -> two 16B reads at +8g and +16+8g
__device__ __forceinline__ v16bf frag_b_lds(const __bf16* Bs, int noff) {
  int lane = threadIdx.x & 31;
  int n = lane & 15, g = lane >> 4;
  const __bf16* p = Bs + (noff + n) * LDSTR + 8 * g;
  uint4 lo = *(const uint4*)(p);
  uint4 hi = *(const uint4*)(p + 16);
  const __bf16* lp = (const __bf16*)&lo;
  const __bf16* hp = (const __bf16*)&hi;
  v16bf r;
#pragma unroll
  for (int i = 0; i < 8; ++i) { r[i] = lp[i]; r[8 + i] = hp[i]; }
  return r;
}

// Cooperative staging: 256 threads move a 128x32 bf16 slab (8KB) per matrix.
// Thread tid: row = tid>>1, half = tid&1 -> 32B (two b128 loads) per slab.
struct Stage { uint4 a0, a1, b0, b1; };

__device__ __forceinline__ Stage stage_load(const __bf16* Ag, int lda,
                                            const __bf16* Bg, int ldb, int k0) {
  int tid = threadIdx.x;
  int row = tid >> 1, half = tid & 1;
  const __bf16* pa = Ag + (size_t)row * lda + k0 + half * 16;
  const __bf16* pb = Bg + (size_t)row * ldb + k0 + half * 16;
  Stage s;
  s.a0 = *(const uint4*)(pa);
  s.a1 = *(const uint4*)(pa + 8);
  s.b0 = *(const uint4*)(pb);
  s.b1 = *(const uint4*)(pb + 8);
  return s;
}

__device__ __forceinline__ void stage_store(const Stage& s, __bf16* As, __bf16* Bs) {
  int tid = threadIdx.x;
  int row = tid >> 1, half = tid & 1;
  __bf16* qa = As + row * LDSTR + half * 16;
  __bf16* qb = Bs + row * LDSTR + half * 16;
  *(uint4*)(qa) = s.a0;
  *(uint4*)(qa + 8) = s.a1;
  *(uint4*)(qb) = s.b0;
  *(uint4*)(qb + 8) = s.b1;
}

// Wave tiling: 8 waves (2 in M x 4 in N); wave = 4x2 tiles of 16x16 = 64x32.
struct TileCtx { int m0, n0, ln, g; };
__device__ __forceinline__ TileCtx tile_ctx(int bm, int bn) {
  int w = threadIdx.x >> 5;
  int lane = threadIdx.x & 31;
  TileCtx t;
  t.m0 = bm + (w >> 2) * 64;
  t.n0 = bn + (w & 3) * 32;
  t.ln = lane & 15;
  t.g  = lane >> 4;
  return t;
}

// Core: acc[4][2] += A(block base, row-major, contract last) x B([N][K] layout)
// with double-buffered LDS staging; global loads for slab k+1 are issued
// before the WMMAs on slab k so VMEM latency overlaps matrix math.
__device__ __forceinline__ void wmma_block(const __bf16* Ag, int lda,
                                           const __bf16* Bg, int ldb,
                                           int K, v8f acc[4][2]) {
  __shared__ __bf16 As[2][128 * LDSTR];
  __shared__ __bf16 Bs[2][128 * LDSTR];
  int w = threadIdx.x >> 5;
  int moff = (w >> 2) * 64;
  int noff = (w & 3) * 32;

  Stage s0 = stage_load(Ag, lda, Bg, ldb, 0);
  stage_store(s0, As[0], Bs[0]);
  __syncthreads();

  for (int k0 = 0; k0 < K; k0 += 32) {
    int buf = (k0 >> 5) & 1;
    bool more = (k0 + 32) < K;
    Stage nx;
    if (more) nx = stage_load(Ag, lda, Bg, ldb, k0 + 32);

    v16bf a[4], b[2];
#pragma unroll
    for (int i = 0; i < 4; ++i) a[i] = frag_a_lds(As[buf], moff + 16 * i);
#pragma unroll
    for (int j = 0; j < 2; ++j) b[j] = frag_b_lds(Bs[buf], noff + 16 * j);
#pragma unroll
    for (int i = 0; i < 4; ++i)
#pragma unroll
      for (int j = 0; j < 2; ++j)
        acc[i][j] = __builtin_amdgcn_wmma_f32_16x16x32_bf16(
            false, a[i], false, b[j], (short)0, acc[i][j], false, false);

    if (more) stage_store(nx, As[buf ^ 1], Bs[buf ^ 1]);
    __syncthreads();
  }
}

// C/D fragment element (i,j,r) -> row = m0+16i+r+8g, col = n0+16j+ln
template <typename F>
__device__ __forceinline__ void epilogue(const TileCtx& t, v8f acc[4][2], F f) {
#pragma unroll
  for (int i = 0; i < 4; ++i)
#pragma unroll
    for (int j = 0; j < 2; ++j)
#pragma unroll
      for (int r = 0; r < 8; ++r) {
        int row = t.m0 + 16 * i + r + 8 * t.g;
        int col = t.n0 + 16 * j + t.ln;
        f(row, col, acc[i][j][r]);
      }
}

// ------------------------------- kernels -----------------------------------

__global__ void k_cvt(const float* __restrict__ src, __bf16* __restrict__ dst, long n) {
  long i = (long)blockIdx.x * blockDim.x + threadIdx.x;
  if (i < n) dst[i] = (__bf16)src[i];
}

// Per-batch fast-weight state init: fp32 master, zero velocity, bf16 mirrors.
__global__ void k_init_fast(const float* __restrict__ W1, const float* __restrict__ W2,
                            const float* __restrict__ W3,
                            float* W1f, float* W2f, float* W3f,
                            float* v1, float* v2, float* v3,
                            __bf16* W1b, __bf16* W2b, __bf16* W2bT, __bf16* W3b) {
  int sel = blockIdx.y;
  long e = (long)blockIdx.x * blockDim.x + threadIdx.x;
  const long NE = (long)HH * DM;
  if (e >= NE) return;
  for (int b = 0; b < BB; ++b) {
    long o = (long)b * NE + e;
    if (sel == 0) {
      float w = W1[e]; W1f[o] = w; v1[o] = 0.f; W1b[o] = (__bf16)w;
    } else if (sel == 1) {
      float w = W2[e]; W2f[o] = w; v2[o] = 0.f; W2b[o] = (__bf16)w;
      int d = (int)(e / HH);                               // W2 is (D,H)
      int h = (int)(e % HH);
      W2bT[(long)b * NE + (long)h * DM + d] = (__bf16)w;   // (H,D) mirror
    } else {
      float w = W3[e]; W3f[o] = w; v3[o] = 0.f; W3b[o] = (__bf16)w;
    }
  }
}

// QKV projection: Y = x @ W^T + bias ; z=0:Q, 1:K (+Kt), 2:V stored negated (+Vt)
__global__ void k_qkv(const __bf16* __restrict__ xb,
                      const __bf16* __restrict__ Wqb, const __bf16* __restrict__ Wkb,
                      const __bf16* __restrict__ Wvb,
                      const float* __restrict__ bq, const float* __restrict__ bk,
                      const float* __restrict__ bv,
                      __bf16* Qb, __bf16* Kb, __bf16* Vn,
                      __bf16* Kt, __bf16* Vt) {
  int z = blockIdx.z;
  const __bf16* W = (z == 0) ? Wqb : (z == 1) ? Wkb : Wvb;
  const float* bias = (z == 0) ? bq : (z == 1) ? bk : bv;
  __bf16* Y = (z == 0) ? Qb : (z == 1) ? Kb : Vn;
  float sgn = (z == 2) ? -1.f : 1.f;
  int bm = blockIdx.x * 128, bn = blockIdx.y * 128;
  TileCtx t = tile_ctx(bm, bn);
  v8f acc[4][2] = {};
  wmma_block(xb + (size_t)bm * DM, DM, W + (size_t)bn * DM, DM, DM, acc);
  epilogue(t, acc, [&](int row, int col, float gval) {
    float v = sgn * (gval + bias[col]);
    Y[(size_t)row * DM + col] = (__bf16)v;
    if (z == 1) {
      int b = row >> 12;
      int tt = row & (TT - 1);
      Kt[((size_t)b * DM + col) * TT + tt] = (__bf16)v;
    } else if (z == 2) {
      int b = row >> 12;
      int tt = row & (TT - 1);
      Vt[((size_t)b * DM + col) * TT + tt] = (__bf16)v;
    }
  });
}

// X_ch @ {W1,W3}^T -> fp32 (C,H). z = batch*2 + sel(0:W1->OA, 1:W3->OB)
__global__ void k_chw(const __bf16* __restrict__ X,
                      const __bf16* __restrict__ W1b, const __bf16* __restrict__ W3b,
                      float* OA, float* OB, int c0) {
  int b = blockIdx.z >> 1, sel = blockIdx.z & 1;
  const __bf16* W = (sel ? W3b : W1b) + (size_t)b * HH * DM;
  float* O = (sel ? OB : OA) + (size_t)b * CC * HH;
  const __bf16* A = X + ((size_t)b * TT + c0) * DM;
  int bm = blockIdx.x * 128, bn = blockIdx.y * 128;
  TileCtx t = tile_ctx(bm, bn);
  v8f acc[4][2] = {};
  wmma_block(A + (size_t)bm * DM, DM, W + (size_t)bn * DM, DM, DM, acc);
  epilogue(t, acc, [&](int row, int col, float gval) {
    O[(size_t)row * HH + col] = gval;
  });
}

// U = silu(A)*Bm -> bf16; transpose=1 writes (H,C) [for gW2 B-operand]
__global__ void k_gate(const float* __restrict__ Ain, const float* __restrict__ Bin,
                       __bf16* Out, int transpose) {
  int b = blockIdx.y;
  long i = (long)blockIdx.x * blockDim.x + threadIdx.x;
  if (i >= (long)CC * HH) return;
  const float* A = Ain + (size_t)b * CC * HH;
  const float* Bm = Bin + (size_t)b * CC * HH;
  float a = A[i], bm = Bm[i];
  float s = a / (1.f + __expf(-a));
  float u = s * bm;
  if (transpose) {
    int c = (int)(i / HH);
    int h = (int)(i % HH);
    Out[(size_t)b * CC * HH + (size_t)h * CC + c] = (__bf16)u;
  } else {
    Out[(size_t)b * CC * HH + i] = (__bf16)u;
  }
}

// gU = (-V)_ch @ W2f : A = Vn rows, B = W2bfT (H,D) -> fp32 (C,H)
__global__ void k_gu(const __bf16* __restrict__ Vn, const __bf16* __restrict__ W2bT,
                     float* gU, int c0) {
  int b = blockIdx.z;
  const __bf16* A = Vn + ((size_t)b * TT + c0) * DM;
  const __bf16* Bb = W2bT + (size_t)b * HH * DM;
  float* O = gU + (size_t)b * CC * HH;
  int bm = blockIdx.x * 128, bn = blockIdx.y * 128;
  TileCtx t = tile_ctx(bm, bn);
  v8f acc[4][2] = {};
  wmma_block(A + (size_t)bm * DM, DM, Bb + (size_t)bn * DM, DM, DM, acc);
  epilogue(t, acc, [&](int row, int col, float gval) {
    O[(size_t)row * HH + col] = gval;
  });
}

// gW2[d,h] = sum_c (-V)^T[d,c] * U^T[h,c] ; fused momentum update of W2.
__global__ void k_gw2(const __bf16* __restrict__ Vt, const __bf16* __restrict__ Ut,
                      float* W2f, float* v2, __bf16* W2b, __bf16* W2bT, int c0) {
  int b = blockIdx.z;
  const __bf16* A = Vt + (size_t)b * DM * TT + c0;  // [d][t], lda=TT
  const __bf16* Bb = Ut + (size_t)b * CC * HH;      // [h][c], ldb=CC
  int bm = blockIdx.x * 128, bn = blockIdx.y * 128;
  TileCtx t = tile_ctx(bm, bn);
  v8f acc[4][2] = {};
  wmma_block(A + (size_t)bm * TT, TT, Bb + (size_t)bn * CC, CC, CC, acc);
  size_t wb = (size_t)b * DM * HH;
  epilogue(t, acc, [&](int row, int col, float gval) {
    size_t e = wb + (size_t)row * HH + col;
    float v = MU * v2[e] + gval;
    float w = W2f[e] - v;
    v2[e] = v; W2f[e] = w;
    W2b[e] = (__bf16)w;
    W2bT[wb + (size_t)col * DM + row] = (__bf16)w;
  });
}

// Backward elementwise: gA,gB written transposed (H,C) bf16.
__global__ void k_bw(const float* __restrict__ Ain, const float* __restrict__ Bin,
                     const float* __restrict__ gUin, __bf16* gAT, __bf16* gBT) {
  int b = blockIdx.y;
  long i = (long)blockIdx.x * blockDim.x + threadIdx.x;
  if (i >= (long)CC * HH) return;
  size_t o = (size_t)b * CC * HH;
  float a = Ain[o + i], bm = Bin[o + i], gu = gUin[o + i];
  float sig = 1.f / (1.f + __expf(-a));
  float s = sig * a;
  float gs = gu * bm;
  float gb = gu * s;
  float ga = gs * (sig * (1.f + a * (1.f - sig)));
  int c = (int)(i / HH);
  int h = (int)(i % HH);
  gAT[o + (size_t)h * CC + c] = (__bf16)ga;
  gBT[o + (size_t)h * CC + c] = (__bf16)gb;
}

// gW1/gW3[h,d] = sum_c gA^T[h,c] * K^T[d,c] ; fused momentum update.
__global__ void k_gw13(const __bf16* __restrict__ gAT, const __bf16* __restrict__ gBT,
                       const __bf16* __restrict__ Kt,
                       float* W1f, float* W3f, float* v1, float* v3,
                       __bf16* W1b, __bf16* W3b, int c0) {
  int b = blockIdx.z >> 1, sel = blockIdx.z & 1;
  const __bf16* A = (sel ? gBT : gAT) + (size_t)b * CC * HH;  // [h][c], lda=CC
  const __bf16* Bb = Kt + (size_t)b * DM * TT + c0;           // [d][t], ldb=TT
  float* Wf = sel ? W3f : W1f;
  float* vv = sel ? v3 : v1;
  __bf16* Wb = sel ? W3b : W1b;
  int bm = blockIdx.x * 128, bn = blockIdx.y * 128;
  TileCtx t = tile_ctx(bm, bn);
  v8f acc[4][2] = {};
  wmma_block(A + (size_t)bm * CC, CC, Bb + (size_t)bn * TT, TT, CC, acc);
  size_t wb = (size_t)b * HH * DM;
  epilogue(t, acc, [&](int row, int col, float gval) {
    size_t e = wb + (size_t)row * DM + col;
    float v = MU * vv[e] + gval;
    float w = Wf[e] - v;
    vv[e] = v; Wf[e] = w;
    Wb[e] = (__bf16)w;
  });
}

// Oq = G @ W2f^T -> fp32 output slab.
__global__ void k_out(const __bf16* __restrict__ Gb, const __bf16* __restrict__ W2b,
                      float* out, int c0) {
  int b = blockIdx.z;
  const __bf16* A = Gb + (size_t)b * CC * HH;        // (C,H), lda=HH
  const __bf16* Bb = W2b + (size_t)b * DM * HH;      // [d][h], ldb=HH
  int bm = blockIdx.x * 128, bn = blockIdx.y * 128;
  TileCtx t = tile_ctx(bm, bn);
  v8f acc[4][2] = {};
  wmma_block(A + (size_t)bm * HH, HH, Bb + (size_t)bn * HH, HH, HH, acc);
  epilogue(t, acc, [&](int row, int col, float gval) {
    out[((size_t)b * TT + c0 + row) * DM + col] = gval;
  });
}

// ------------------------------ launcher ------------------------------------

extern "C" void kernel_launch(void* const* d_in, const int* in_sizes, int n_in,
                              void* d_out, int out_size, void* d_ws, size_t ws_size,
                              hipStream_t stream) {
  const float* x  = (const float*)d_in[0];
  const float* W1 = (const float*)d_in[1];
  const float* W2 = (const float*)d_in[2];
  const float* W3 = (const float*)d_in[3];
  const float* Wq = (const float*)d_in[4];
  const float* bq = (const float*)d_in[5];
  const float* Wk = (const float*)d_in[6];
  const float* bk = (const float*)d_in[7];
  const float* Wv = (const float*)d_in[8];
  const float* bv = (const float*)d_in[9];
  float* out = (float*)d_out;

  char* p = (char*)d_ws;
  auto alloc = [&](size_t bytes) -> char* {
    char* r = p;
    p += (bytes + 255) & ~(size_t)255;
    return r;
  };
  const size_t NXT = (size_t)BB * TT * DM;   // 16.7M elems
  const size_t NW  = (size_t)HH * DM;        // 2M elems
  const size_t NCH = (size_t)CC * HH;        // 1M elems

  __bf16* xb   = (__bf16*)alloc(NXT * 2);
  __bf16* Wqb  = (__bf16*)alloc((size_t)DM * DM * 2);
  __bf16* Wkb  = (__bf16*)alloc((size_t)DM * DM * 2);
  __bf16* Wvb  = (__bf16*)alloc((size_t)DM * DM * 2);
  __bf16* Qb   = (__bf16*)alloc(NXT * 2);
  __bf16* Kb   = (__bf16*)alloc(NXT * 2);
  __bf16* Vn   = (__bf16*)alloc(NXT * 2);
  __bf16* Kt   = (__bf16*)alloc(NXT * 2);
  __bf16* Vt   = (__bf16*)alloc(NXT * 2);
  float*  W1f  = (float*)alloc(BB * NW * 4);
  float*  W2f  = (float*)alloc(BB * NW * 4);
  float*  W3f  = (float*)alloc(BB * NW * 4);
  float*  v1   = (float*)alloc(BB * NW * 4);
  float*  v2   = (float*)alloc(BB * NW * 4);
  float*  v3   = (float*)alloc(BB * NW * 4);
  __bf16* W1b  = (__bf16*)alloc(BB * NW * 2);
  __bf16* W3b  = (__bf16*)alloc(BB * NW * 2);
  __bf16* W2b  = (__bf16*)alloc(BB * NW * 2);
  __bf16* W2bT = (__bf16*)alloc(BB * NW * 2);
  float*  Abuf = (float*)alloc(BB * NCH * 4);
  float*  Bbuf = (float*)alloc(BB * NCH * 4);
  float*  gU   = (float*)alloc(BB * NCH * 4);  // also reused as Gq (bf16 fits)
  __bf16* Ut   = (__bf16*)alloc(BB * NCH * 2);
  __bf16* gAT  = (__bf16*)alloc(BB * NCH * 2);
  __bf16* gBT  = (__bf16*)alloc(BB * NCH * 2);
  __bf16* Gb   = (__bf16*)gU;                  // reuse after k_bw consumed gU

  // ---- input conversions to bf16 ----
  k_cvt<<<dim3((unsigned)(NXT / 256)), 256, 0, stream>>>(x, xb, (long)NXT);
  k_cvt<<<dim3((unsigned)(DM * DM / 256)), 256, 0, stream>>>(Wq, Wqb, (long)DM * DM);
  k_cvt<<<dim3((unsigned)(DM * DM / 256)), 256, 0, stream>>>(Wk, Wkb, (long)DM * DM);
  k_cvt<<<dim3((unsigned)(DM * DM / 256)), 256, 0, stream>>>(Wv, Wvb, (long)DM * DM);

  // ---- fast-weight state init (per batch) ----
  k_init_fast<<<dim3((unsigned)(NW / 256), 3), 256, 0, stream>>>(
      W1, W2, W3, W1f, W2f, W3f, v1, v2, v3, W1b, W2b, W2bT, W3b);

  // ---- QKV projections ----
  k_qkv<<<dim3(BB * TT / 128, DM / 128, 3), 256, 0, stream>>>(
      xb, Wqb, Wkb, Wvb, bq, bk, bv, Qb, Kb, Vn, Kt, Vt);

  // ---- fast-weight scan over chunks ----
  for (int nc = 0; nc < NCHUNK; ++nc) {
    int c0 = nc * CC;
    // A = K@W1^T, Bm = K@W3^T
    k_chw<<<dim3(CC / 128, HH / 128, BB * 2), 256, 0, stream>>>(
        Kb, W1b, W3b, Abuf, Bbuf, c0);
    // U^T = (silu(A)*Bm)^T
    k_gate<<<dim3((unsigned)(NCH / 256), BB), 256, 0, stream>>>(Abuf, Bbuf, Ut, 1);
    // gU = (-V)@W2f (old W2)
    k_gu<<<dim3(CC / 128, HH / 128, BB), 256, 0, stream>>>(Vn, W2bT, gU, c0);
    // gW2 + W2 momentum update (also refreshes bf16 mirrors)
    k_gw2<<<dim3(DM / 128, HH / 128, BB), 256, 0, stream>>>(
        Vt, Ut, W2f, v2, W2b, W2bT, c0);
    // backward elementwise -> gA^T, gB^T
    k_bw<<<dim3((unsigned)(NCH / 256), BB), 256, 0, stream>>>(Abuf, Bbuf, gU, gAT, gBT);
    // gW1/gW3 + W1/W3 momentum updates
    k_gw13<<<dim3(HH / 128, DM / 128, BB * 2), 256, 0, stream>>>(
        gAT, gBT, Kt, W1f, W3f, v1, v3, W1b, W3b, c0);
    // apply updated weights to queries: Aq, Bq (reuse Abuf/Bbuf)
    k_chw<<<dim3(CC / 128, HH / 128, BB * 2), 256, 0, stream>>>(
        Qb, W1b, W3b, Abuf, Bbuf, c0);
    // Gq = silu(Aq)*Bq (reuse gU buffer as bf16)
    k_gate<<<dim3((unsigned)(NCH / 256), BB), 256, 0, stream>>>(Abuf, Bbuf, Gb, 0);
    // Oq = Gq @ W2f^T -> output
    k_out<<<dim3(CC / 128, DM / 128, BB), 256, 0, stream>>>(Gb, W2b, out, c0);
  }
}
